// MultiHeadAttentionKVCache_33354716020800
// MI455X (gfx1250) — matrix operand order repaired and checked by reading, not compiled, not run on hardware
//
#include <hip/hip_runtime.h>
#include <hip/hip_bf16.h>

#define Bd 2
#define Sd 1024
#define Dd 2048
#define CACHE_LEN 4096
#define Hd 32
#define KVHd 8
#define HDd 64
#define POSd 3072
#define Td 4096   /* POS + S */
#define REPd 4

typedef __bf16 bf16;
typedef __attribute__((ext_vector_type(16))) __bf16 v16bf;
typedef __attribute__((ext_vector_type(8)))  __bf16 v8bf;
typedef __attribute__((ext_vector_type(8)))  float  v8f;
typedef __attribute__((ext_vector_type(4)))  int    v4i;

#define AS1 __attribute__((address_space(1)))
#define AS3 __attribute__((address_space(3)))

// 16-byte global -> LDS copy, async on CDNA5 when available.
__device__ __forceinline__ void async_copy16(const bf16* g, bf16* l) {
#if defined(__has_builtin) && __has_builtin(__builtin_amdgcn_global_load_async_to_lds_b128)
  bf16* gnc = const_cast<bf16*>(g);
  __builtin_amdgcn_global_load_async_to_lds_b128((AS1 v4i*)gnc, (AS3 v4i*)l, 0, 0);
#else
  *(v8bf*)l = *(const v8bf*)g;
#endif
}

__device__ __forceinline__ void wait_async0() {
#if defined(__has_builtin) && __has_builtin(__builtin_amdgcn_s_wait_asynccnt)
  __builtin_amdgcn_s_wait_asynccnt(0);
#else
  asm volatile("s_wait_asynccnt 0" ::: "memory");
#endif
}

// ---------------- elementwise f32 -> bf16 ----------------
__global__ void f32_to_bf16_kernel(const float* __restrict__ src,
                                   bf16* __restrict__ dst, int n) {
  int i = blockIdx.x * blockDim.x + threadIdx.x;
  if (i < n) dst[i] = (bf16)src[i];
}

// ------------- transpose + convert: W (K x N) f32 -> WT (N x K) bf16 -------------
__global__ void transpose_bf16_kernel(const float* __restrict__ W,
                                      bf16* __restrict__ WT, int K, int N) {
  int i = blockIdx.x * blockDim.x + threadIdx.x;   // dst-linear (coalesced writes)
  if (i >= N * K) return;
  int n = i / K, k = i - n * K;
  WT[i] = (bf16)W[k * N + n];
}

// --- splice old cache rows [0,POS) into combined bf16 buffers (V transposed) ---
__global__ void cache_convert_kernel(const float* __restrict__ kc,
                                     const float* __restrict__ vc,
                                     bf16* __restrict__ ckb,
                                     bf16* __restrict__ cvbT) {
  int i = blockIdx.x * blockDim.x + threadIdx.x;   // over B*POS*KVH*HD, src-linear
  if (i >= Bd * POSd * KVHd * HDd) return;
  int hd  = i & 63;
  int g   = (i >> 6) & 7;
  int rem = i >> 9;            // b*POS + t
  int t   = rem % POSd;
  int b   = rem / POSd;
  size_t src = (((size_t)b * CACHE_LEN + t) * KVHd + g) * HDd + hd;
  ckb [(((size_t)b * KVHd + g) * Td + t) * HDd + hd] = (bf16)kc[src];
  cvbT[(((size_t)b * KVHd + g) * HDd + hd) * Td + t] = (bf16)vc[src];
}

// ---------------- generic bf16 WMMA GEMM: C = A(MxK) * BT(NxK)^T + bias ----------------
// Block: 128 threads = 4 waves; block tile 128x64; wave tile 32x64.
// B tile staged to LDS via async copies (shared by all 4 waves), double buffered;
// K loop unrolled x2 with explicit ping-pong (no register copies).
// mode 0: store bf16 Q*scale -> qb; mode 1: K -> ckb; mode 2: V -> cvbT; mode 3: f32 out.
__global__ __launch_bounds__(128)
void gemm_wmma_kernel(const bf16* __restrict__ A, const bf16* __restrict__ BT,
                      const float* __restrict__ bias, void* __restrict__ dstv,
                      int Ndim, int Kdim, int mode) {
  __shared__ __align__(32) bf16 btile[2][64][32];
  const int tid  = threadIdx.x;
  const int lane = tid & 31;
  const int wave = tid >> 5;
  const int half = lane >> 4;
  const int l16  = lane & 15;
  const int m0   = blockIdx.y * 128 + wave * 32;
  const int n0   = blockIdx.x * 64;

  const int aklo  = half ? 16 : 0;    // A-frag K offsets, contiguous k = 16*half + s
  const int akhi  = half ? 24 : 8;
  const int bkoff = half ? 16 : 0;    // B-frag K offset

  v8f acc[2][4] = {};
  v16bf a_cur[2], a_nxt[2];

  auto stageB = [&](int kk, int buf) {
#pragma unroll
    for (int c = 0; c < 2; c++) {
      int ch = tid + c * 128;          // 256 x 16B chunks = 64 rows x 64B
      int n  = ch >> 2;
      int k8 = (ch & 3) * 8;
      async_copy16(BT + (size_t)(n0 + n) * Kdim + kk + k8, &btile[buf][n][k8]);
    }
  };
  auto loadA = [&](int kk, v16bf* out) {
#pragma unroll
    for (int mt = 0; mt < 2; mt++) {
      const bf16* ap = A + (size_t)(m0 + mt * 16 + l16) * Kdim + kk;
      v8bf lo = *(const v8bf*)(ap + aklo);
      v8bf hi = *(const v8bf*)(ap + akhi);
      out[mt] = __builtin_shufflevector(lo, hi, 0,1,2,3,4,5,6,7,8,9,10,11,12,13,14,15);
    }
  };
  // preload all B frags, single dscnt wait, then 8 back-to-back WMMAs
  auto compute = [&](int buf, const v16bf* a) {
    v16bf bfr[4];
#pragma unroll
    for (int j = 0; j < 4; j++)
      bfr[j] = *(const v16bf*)&btile[buf][j * 16 + l16][bkoff];
#pragma unroll
    for (int j = 0; j < 4; j++)
#pragma unroll
      for (int mt = 0; mt < 2; mt++)
        acc[mt][j] = __builtin_amdgcn_wmma_f32_16x16x32_bf16(false, a[mt], false, bfr[j],
                                                             (short)0, acc[mt][j], false, false);
  };

  stageB(0, 0);
  loadA(0, a_cur);
  wait_async0();
  __syncthreads();

  for (int kk = 0; kk < Kdim; kk += 64) {          // Kdim is a multiple of 64
    stageB(kk + 32, 1);
    loadA(kk + 32, a_nxt);
    compute(0, a_cur);
    wait_async0();
    __syncthreads();

    const bool more = (kk + 64) < Kdim;
    if (more) { stageB(kk + 64, 0); loadA(kk + 64, a_cur); }
    compute(1, a_nxt);
    wait_async0();
    __syncthreads();
  }

#pragma unroll
  for (int mt = 0; mt < 2; mt++) {
#pragma unroll
    for (int j = 0; j < 4; j++) {
      int ncol = n0 + j * 16 + l16;
      float bv = bias[ncol];
#pragma unroll
      for (int r = 0; r < 8; r++) {
        int   mrow = m0 + mt * 16 + half * 8 + r;
        float val  = acc[mt][j][r] + bv;
        if (mode == 3) {
          ((float*)dstv)[(size_t)mrow * Ndim + ncol] = val;
        } else {
          if (mode == 0) val *= 0.125f;        // fold softmax scale into Q
          bf16 bb = (bf16)val;
          int b = mrow / Sd, s = mrow - b * Sd;
          int hd = ncol & 63;
          if (mode == 0) {
            int h = ncol >> 6;
            ((bf16*)dstv)[(((size_t)b * Hd + h) * Sd + s) * HDd + hd] = bb;
          } else if (mode == 1) {
            int g = ncol >> 6;
            ((bf16*)dstv)[(((size_t)b * KVHd + g) * Td + POSd + s) * HDd + hd] = bb;
          } else {
            int g = ncol >> 6;
            ((bf16*)dstv)[(((size_t)b * KVHd + g) * HDd + hd) * Td + POSd + s] = bb;
          }
        }
      }
    }
  }
}

// ---------------- flash-style attention, all-WMMA ----------------
// grid.x = B*H (head), grid.y = S/64. 4 waves x 16 q-rows; K/V tiles staged to
// LDS with async copies (shared by the 4 waves), double buffered, 32 keys/step.
__global__ __launch_bounds__(128)
void attention_wmma_kernel(const bf16* __restrict__ qb, const bf16* __restrict__ ckb,
                           const bf16* __restrict__ cvbT, bf16* __restrict__ zb) {
  __shared__ __align__(32) bf16 ktile[2][32][64];   // [key][hd]
  __shared__ __align__(32) bf16 vtile[2][64][32];   // [hd][key]
  __shared__ __align__(32) bf16 plds[4][16][32];    // per-wave P transpose staging
  const int tid  = threadIdx.x;
  const int lane = tid & 31;
  const int wave = tid >> 5;
  const int half = lane >> 4;
  const int l16  = lane & 15;
  const int hf   = blockIdx.x;         // b*H + h
  const int b    = hf / Hd;
  const int h    = hf % Hd;
  const int g    = h / REPd;
  const int q0   = blockIdx.y * 64 + wave * 16;

  const bf16* qbase = qb   + ((size_t)(b * Hd   + h) * Sd)  * HDd;
  const bf16* kbase = ckb  + ((size_t)(b * KVHd + g) * Td)  * HDd;
  const bf16* vbase = cvbT + ((size_t)(b * KVHd + g) * HDd) * Td;

  const int aklo  = half ? 16 : 0;
  const int akhi  = half ? 24 : 8;
  const int bkoff = half ? 16 : 0;

  auto stageKV = [&](int t0, int buf) {
#pragma unroll
    for (int c = 0; c < 2; c++) {
      int ch = tid + c * 128;                       // 256 x 16B chunks each
      int kr = ch >> 3, k8 = (ch & 7) * 8;          // K: 32 rows x 128B
      async_copy16(kbase + (size_t)(t0 + kr) * HDd + k8, &ktile[buf][kr][k8]);
      int vr = ch >> 2, t8 = (ch & 3) * 8;          // V: 64 rows x 64B
      async_copy16(vbase + (size_t)vr * Td + t0 + t8, &vtile[buf][vr][t8]);
    }
  };

  // Q fragments (A layout), K-dim = hd in 2 chunks of 32
  v16bf qf[2];
#pragma unroll
  for (int c = 0; c < 2; c++) {
    v8bf lo = *(const v8bf*)(qbase + (size_t)(q0 + l16) * HDd + c * 32 + aklo);
    v8bf hi = *(const v8bf*)(qbase + (size_t)(q0 + l16) * HDd + c * 32 + akhi);
    qf[c] = __builtin_shufflevector(lo, hi, 0,1,2,3,4,5,6,7,8,9,10,11,12,13,14,15);
  }

  v8f acc[4] = {};
  float mrow[8], lrow[8];
#pragma unroll
  for (int r = 0; r < 8; r++) { mrow[r] = -1e30f; lrow[r] = 0.f; }

  stageKV(0, 0);
  wait_async0();
  __syncthreads();

  for (int t0 = 0; t0 < Td; t0 += 32) {
    const int buf = (t0 >> 5) & 1;
    if (t0 + 32 < Td) stageKV(t0 + 32, buf ^ 1);

    // ---- scores: Q(16x64) . K^T(64x32); preload 4 K-frags, then 4 WMMAs ----
    v16bf kf[2][2];
#pragma unroll
    for (int nt = 0; nt < 2; nt++)
#pragma unroll
      for (int ch = 0; ch < 2; ch++)
        kf[nt][ch] = *(const v16bf*)&ktile[buf][nt * 16 + l16][ch * 32 + bkoff];
    v8f sc[2];
#pragma unroll
    for (int nt = 0; nt < 2; nt++) {
      v8f c = {};
      c = __builtin_amdgcn_wmma_f32_16x16x32_bf16(false, qf[0], false, kf[nt][0],
                                                  (short)0, c, false, false);
      c = __builtin_amdgcn_wmma_f32_16x16x32_bf16(false, qf[1], false, kf[nt][1],
                                                  (short)0, c, false, false);
      sc[nt] = c;
    }

    // ---- online softmax (row = half*8 + r, cols across 16-lane half) ----
#pragma unroll
    for (int r = 0; r < 8; r++) {
      float s0 = sc[0][r], s1 = sc[1][r];
      float mx = fmaxf(s0, s1);
#pragma unroll
      for (int mk = 1; mk <= 8; mk <<= 1) mx = fmaxf(mx, __shfl_xor(mx, mk, 32));
      float mnew = fmaxf(mrow[r], mx);
      float corr = __expf(mrow[r] - mnew);
      float p0 = __expf(s0 - mnew);
      float p1 = __expf(s1 - mnew);
      float ps = p0 + p1;
#pragma unroll
      for (int mk = 1; mk <= 8; mk <<= 1) ps += __shfl_xor(ps, mk, 32);
      lrow[r] = lrow[r] * corr + ps;
      mrow[r] = mnew;
#pragma unroll
      for (int j = 0; j < 4; j++) acc[j][r] *= corr;
      int row = half * 8 + r;
      plds[wave][row][l16]      = (bf16)p0;
      plds[wave][row][16 + l16] = (bf16)p1;
    }
    asm volatile("s_wait_dscnt 0" ::: "memory");   // per-wave LDS transpose fence
    v8bf plo = *(const v8bf*)&plds[wave][l16][aklo];
    v8bf phi = *(const v8bf*)&plds[wave][l16][akhi];
    v16bf pf = __builtin_shufflevector(plo, phi, 0,1,2,3,4,5,6,7,8,9,10,11,12,13,14,15);

    // ---- O += P(16x32) . V(32x64); preload 4 V-frags, then 4 WMMAs ----
    v16bf vf[4];
#pragma unroll
    for (int j = 0; j < 4; j++)
      vf[j] = *(const v16bf*)&vtile[buf][j * 16 + l16][bkoff];
#pragma unroll
    for (int j = 0; j < 4; j++)
      acc[j] = __builtin_amdgcn_wmma_f32_16x16x32_bf16(false, pf, false, vf[j],
                                                       (short)0, acc[j], false, false);

    wait_async0();
    __syncthreads();
  }

  // ---- normalize and store z (B,S,H*HD) bf16 ----
#pragma unroll
  for (int j = 0; j < 4; j++) {
#pragma unroll
    for (int r = 0; r < 8; r++) {
      int row  = half * 8 + r;
      int srow = q0 + row;
      float o  = acc[j][r] / lrow[r];
      zb[((size_t)(b * Sd + srow)) * (Hd * HDd) + h * HDd + j * 16 + l16] = (bf16)o;
    }
  }
}

extern "C" void kernel_launch(void* const* d_in, const int* in_sizes, int n_in,
                              void* d_out, int out_size, void* d_ws, size_t ws_size,
                              hipStream_t stream) {
  const float* x   = (const float*)d_in[0];
  const float* kc  = (const float*)d_in[1];
  const float* vc  = (const float*)d_in[2];
  const float* Wq  = (const float*)d_in[3];
  const float* bq  = (const float*)d_in[4];
  const float* Wk  = (const float*)d_in[5];
  const float* bk  = (const float*)d_in[6];
  const float* Wv  = (const float*)d_in[7];
  const float* bv  = (const float*)d_in[8];
  const float* Wo  = (const float*)d_in[9];
  const float* bo  = (const float*)d_in[10];

  char* ws = (char*)d_ws;
  size_t off = 0;
  bf16* xb   = (bf16*)(ws + off); off += (size_t)Bd*Sd*Dd       * 2;  // 8MB
  bf16* WqT  = (bf16*)(ws + off); off += (size_t)Dd*Hd*HDd      * 2;  // 8MB
  bf16* WkT  = (bf16*)(ws + off); off += (size_t)Dd*KVHd*HDd    * 2;  // 2MB
  bf16* WvT  = (bf16*)(ws + off); off += (size_t)Dd*KVHd*HDd    * 2;  // 2MB
  bf16* WoT  = (bf16*)(ws + off); off += (size_t)Hd*HDd*Dd      * 2;  // 8MB
  bf16* qb   = (bf16*)(ws + off); off += (size_t)Bd*Hd*Sd*HDd   * 2;  // 8MB
  bf16* ckb  = (bf16*)(ws + off); off += (size_t)Bd*KVHd*Td*HDd * 2;  // 8MB
  bf16* cvbT = (bf16*)(ws + off); off += (size_t)Bd*KVHd*HDd*Td * 2;  // 8MB
  bf16* zb   = (bf16*)(ws + off); off += (size_t)Bd*Sd*Hd*HDd   * 2;  // 8MB

  const int TPB = 256;
  {
    int n = Bd * Sd * Dd;
    f32_to_bf16_kernel<<<(n + TPB - 1) / TPB, TPB, 0, stream>>>(x, xb, n);
  }
  transpose_bf16_kernel<<<(Dd*Hd*HDd   + TPB-1)/TPB, TPB, 0, stream>>>(Wq, WqT, Dd, Hd*HDd);
  transpose_bf16_kernel<<<(Dd*KVHd*HDd + TPB-1)/TPB, TPB, 0, stream>>>(Wk, WkT, Dd, KVHd*HDd);
  transpose_bf16_kernel<<<(Dd*KVHd*HDd + TPB-1)/TPB, TPB, 0, stream>>>(Wv, WvT, Dd, KVHd*HDd);
  transpose_bf16_kernel<<<(Hd*HDd*Dd   + TPB-1)/TPB, TPB, 0, stream>>>(Wo, WoT, Hd*HDd, Dd);
  {
    int n = Bd * POSd * KVHd * HDd;
    cache_convert_kernel<<<(n + TPB - 1) / TPB, TPB, 0, stream>>>(kc, vc, ckb, cvbT);
  }

  // QKV projections (WMMA, 128-row M blocks)
  gemm_wmma_kernel<<<dim3((Hd*HDd)/64,  (Bd*Sd)/128), 128, 0, stream>>>(xb, WqT, bq, qb,   Hd*HDd,   Dd, 0);
  gemm_wmma_kernel<<<dim3((KVHd*HDd)/64,(Bd*Sd)/128), 128, 0, stream>>>(xb, WkT, bk, ckb,  KVHd*HDd, Dd, 1);
  gemm_wmma_kernel<<<dim3((KVHd*HDd)/64,(Bd*Sd)/128), 128, 0, stream>>>(xb, WvT, bv, cvbT, KVHd*HDd, Dd, 2);

  // attention (WMMA, online softmax, async K/V staging)
  attention_wmma_kernel<<<dim3(Bd*Hd, Sd/64), 128, 0, stream>>>(qb, ckb, cvbT, zb);

  // output projection -> d_out (f32)
  gemm_wmma_kernel<<<dim3(Dd/64, (Bd*Sd)/128), 128, 0, stream>>>(zb, WoT, bo, d_out, Dd, Hd*HDd, 3);

  (void)in_sizes; (void)n_in; (void)out_size; (void)ws_size;
}
